// ExpandingLinear_75720273428633
// MI455X (gfx1250) — matrix operand order repaired and checked by reading, not compiled, not run on hardware
//
#include <hip/hip_runtime.h>
#include <hip/hip_bf16.h>

// Problem constants (match reference)
#define OUT_F   2048
#define IN_BASE 2048
#define N_EMB   64
#define IN_TOT  2176   // 2048 + 2*64
#define BATCH   8192

#define TILE_M 128
#define TILE_N 128
#define TILE_K 32
#define LDT    40              // padded LDS row stride (bf16) -> conflict-free b128 reads
#define NKT    (IN_TOT / TILE_K)   // 68

typedef __attribute__((ext_vector_type(16))) __bf16 v16bf;
typedef __attribute__((ext_vector_type(8)))  __bf16 v8bf;
typedef __attribute__((ext_vector_type(4)))  __bf16 v4bf;
typedef __attribute__((ext_vector_type(8)))  float  v8f;
typedef int vsi4 __attribute__((vector_size(16)));   // matches builtin's V4i pointee

#if defined(__gfx1250__) && __has_builtin(__builtin_amdgcn_global_load_async_to_lds_b128)
#define HAVE_ASYNC_LDS 1
#endif

// ---------- async global->LDS copy of 16 bytes (8 bf16), with sync fallback ----------
static __device__ __forceinline__ void async_copy16(const __bf16* g, __bf16* l) {
#ifdef HAVE_ASYNC_LDS
    __builtin_amdgcn_global_load_async_to_lds_b128(
        (__attribute__((address_space(1))) vsi4*)g,
        (__attribute__((address_space(3))) vsi4*)l, 0, 0);
#else
    *(v8bf*)l = *(const v8bf*)g;
#endif
}
template <int N>
static __device__ __forceinline__ void wait_async() {
#ifdef HAVE_ASYNC_LDS
#if __has_builtin(__builtin_amdgcn_s_wait_asynccnt)
    __builtin_amdgcn_s_wait_asynccnt(N);
#else
    asm volatile("s_wait_asynccnt %0" ::"i"(N));
#endif
#endif
}

// ---------- bf16 helpers (round-to-nearest-even split) ----------
static __device__ __forceinline__ __bf16 bf_rn(float f) {
    unsigned u = __builtin_bit_cast(unsigned, f);
    unsigned r = u + 0x7FFFu + ((u >> 16) & 1u);
    unsigned short s = (unsigned short)(r >> 16);
    return __builtin_bit_cast(__bf16, s);
}
static __device__ __forceinline__ float bf_f32(__bf16 b) {
    unsigned short s = __builtin_bit_cast(unsigned short, b);
    return __builtin_bit_cast(float, ((unsigned)s) << 16);
}
static __device__ __forceinline__ void split4(float4 f, v4bf& h, v4bf& l) {
    float fs[4] = {f.x, f.y, f.z, f.w};
#pragma unroll
    for (int i = 0; i < 4; ++i) {
        __bf16 hi = bf_rn(fs[i]);
        h[i] = hi;
        l[i] = bf_rn(fs[i] - bf_f32(hi));
    }
}
// Load 16 bf16 as two 8-element (16B-aligned) LDS chunks at p and p+stride8.
static __device__ __forceinline__ v16bf ld16_pair(const __bf16* p, int stride8) {
    v8bf a = *(const v8bf*)p;
    v8bf b = *(const v8bf*)(p + stride8);
    return __builtin_shufflevector(a, b, 0, 1, 2, 3, 4, 5, 6, 7,
                                         8, 9, 10, 11, 12, 13, 14, 15);
}

// ---------- Kernel 1: zero dense W and bias ----------
__global__ void zero_kernel(float* __restrict__ p, int n) {
    int i = blockIdx.x * blockDim.x + threadIdx.x;
    int stride = gridDim.x * blockDim.x;
    for (; i < n; i += stride) p[i] = 0.0f;
}

// ---------- Kernel 2: COO scatter-add into dense W ----------
__global__ void scatter_w_kernel(float* __restrict__ W, const float* __restrict__ v,
                                 const int* __restrict__ r, const int* __restrict__ c, int n) {
    int i = blockIdx.x * blockDim.x + threadIdx.x;
    if (i < n) atomicAdd(&W[(size_t)r[i] * IN_TOT + c[i]], v[i]);
}

// ---------- Kernel 3: bias scatter-add ----------
__global__ void scatter_b_kernel(float* __restrict__ b, const float* __restrict__ v,
                                 const int* __restrict__ idx, int n) {
    int i = blockIdx.x * blockDim.x + threadIdx.x;
    if (i < n) atomicAdd(&b[idx[i]], v[i]);
}

// ---------- Kernel 4: expanded embedding features xe[B,128] ----------
__global__ void embed_kernel(const float* __restrict__ x, float* __restrict__ xe,
                             const float* __restrict__ e0v, const int* __restrict__ e0p,
                             const float* __restrict__ e1v, const int* __restrict__ e1p) {
    __shared__ float sv0[N_EMB]; __shared__ int sp0[N_EMB];
    __shared__ float sv1[N_EMB]; __shared__ int sp1[N_EMB];
    if (threadIdx.x < N_EMB) {
        sv0[threadIdx.x] = e0v[threadIdx.x]; sp0[threadIdx.x] = e0p[threadIdx.x];
        sv1[threadIdx.x] = e1v[threadIdx.x]; sp1[threadIdx.x] = e1p[threadIdx.x];
    }
    __syncthreads();
    int row = blockIdx.x * blockDim.x + threadIdx.x;
    if (row >= BATCH) return;
    const float* xr = x + (size_t)row * IN_BASE;
    float* er = xe + (size_t)row * (2 * N_EMB);
    for (int c = 0; c < N_EMB; ++c)       // block 0: parents in [0, 2048)
        er[c] = sv0[c] * xr[sp0[c]];
    for (int c = 0; c < N_EMB; ++c) {     // block 1: parents in [0, 2112)
        int p = sp1[c];
        float s = (p < IN_BASE) ? xr[p] : er[p - IN_BASE];
        er[N_EMB + c] = sv1[c] * s;
    }
}

// ---------- Kernel 5a: split Xext (x | xe) into bf16 hi/lo planes [B, 2176] ----------
__global__ void split_x_kernel(const float* __restrict__ x, const float* __restrict__ xe,
                               __bf16* __restrict__ Xhi, __bf16* __restrict__ Xlo) {
    const int idx = blockIdx.x * blockDim.x + threadIdx.x;   // float4-chunk id
    const int row = idx / (IN_TOT / 4);
    const int c4  = (idx % (IN_TOT / 4)) * 4;
    if (row >= BATCH) return;
    const float* src = (c4 < IN_BASE)
        ? x  + (size_t)row * IN_BASE     + c4
        : xe + (size_t)row * (2 * N_EMB) + (c4 - IN_BASE);
    float4 f = *(const float4*)src;
    v4bf h, l;
    split4(f, h, l);
    *(v4bf*)(Xhi + (size_t)row * IN_TOT + c4) = h;
    *(v4bf*)(Xlo + (size_t)row * IN_TOT + c4) = l;
}

// ---------- Kernel 5b: split dense W into bf16 hi/lo planes [OUT, 2176] ----------
__global__ void split_w_kernel(const float* __restrict__ W,
                               __bf16* __restrict__ Whi, __bf16* __restrict__ Wlo) {
    const int idx = blockIdx.x * blockDim.x + threadIdx.x;
    const int row = idx / (IN_TOT / 4);
    const int c4  = (idx % (IN_TOT / 4)) * 4;
    if (row >= OUT_F) return;
    float4 f = *(const float4*)(W + (size_t)row * IN_TOT + c4);
    v4bf h, l;
    split4(f, h, l);
    *(v4bf*)(Whi + (size_t)row * IN_TOT + c4) = h;
    *(v4bf*)(Wlo + (size_t)row * IN_TOT + c4) = l;
}

// ---------- Kernel 6: double-buffered async bf16x3 WMMA GEMM  out = Xext @ W^T + b ----------
__global__ __launch_bounds__(256, 2)
void gemm_kernel(const __bf16* __restrict__ Xhi, const __bf16* __restrict__ Xlo,
                 const __bf16* __restrict__ Whi, const __bf16* __restrict__ Wlo,
                 const float* __restrict__ bias, float* __restrict__ out) {
    // 2 buffers x {Ahi, Alo, Bhi, Blo} x 128 rows x LDT  = 80 KB
    __shared__ __bf16 smem[2][4][TILE_M * LDT];

    const int nBase = blockIdx.x * TILE_N;
    const int mBase = blockIdx.y * TILE_M;
    const int tid   = threadIdx.x;
    const int lane  = tid & 31;
    const int wave  = tid >> 5;
    const int waveM = wave & 1;   // 2 wave rows (64 M each)
    const int waveN = wave >> 1;  // 4 wave cols (32 N each)
    const int lrow  = lane & 15;
    const int hiHalf = lane >> 4;

    // staging: each thread owns 2 rows x one 8-elem chunk per plane per tile
    const int row0 = tid >> 2;        // 0..63
    const int kch  = (tid & 3) << 3;  // 0,8,16,24

    v8f acc[4][2];
#pragma unroll
    for (int mi = 0; mi < 4; ++mi)
#pragma unroll
        for (int ni = 0; ni < 2; ++ni)
#pragma unroll
            for (int r = 0; r < 8; ++r) acc[mi][ni][r] = 0.0f;

    auto issue = [&](int t, int buf) {
        const size_t kb = (size_t)t * TILE_K;
#pragma unroll
        for (int p = 0; p < 2; ++p) {
            const int row = row0 + p * 64;
            const size_t gA = (size_t)(mBase + row) * IN_TOT + kb + kch;
            const size_t gB = (size_t)(nBase + row) * IN_TOT + kb + kch;
            const int l = row * LDT + kch;
            async_copy16(Xhi + gA, &smem[buf][0][l]);
            async_copy16(Xlo + gA, &smem[buf][1][l]);
            async_copy16(Whi + gB, &smem[buf][2][l]);
            async_copy16(Wlo + gB, &smem[buf][3][l]);
        }
    };

    issue(0, 0);
    for (int t = 0; t < NKT; ++t) {
        const int cur = t & 1;
        if (t + 1 < NKT) {
            issue(t + 1, cur ^ 1);  // overlap next tile's DMA with this tile's math
            wait_async<8>();        // FIFO: current tile's 8 loads are complete
        } else {
            wait_async<0>();
        }
        __syncthreads();

        // ---- fragment loads per CDNA5 WMMA VGPR layouts ----
        v16bf ahi[4], alo[4], bhi[2], blo[2];
#pragma unroll
        for (int mi = 0; mi < 4; ++mi) {
            // A 16x32 bf16: lanes<16 hold K{0-7,16-23}; lanes>=16 hold K{8-15,24-31}
            const int row = waveM * 64 + mi * 16 + lrow;
            const int kc  = hiHalf ? 8 : 0;
            ahi[mi] = ld16_pair(&smem[cur][0][row * LDT + kc], 16);
            alo[mi] = ld16_pair(&smem[cur][1][row * LDT + kc], 16);
        }
#pragma unroll
        for (int ni = 0; ni < 2; ++ni) {
            // B 32x16 bf16: lanes<16 hold K{0-7,16-23} of col n=lane; lanes>=16 hold K{8-15,24-31}
            const int row = waveN * 32 + ni * 16 + lrow;
            const int kc  = hiHalf ? 8 : 0;
            bhi[ni] = ld16_pair(&smem[cur][2][row * LDT + kc], 16);
            blo[ni] = ld16_pair(&smem[cur][3][row * LDT + kc], 16);
        }

        // ---- bf16x3 split accumulation: hi*hi + hi*lo + lo*hi ----
#pragma unroll
        for (int mi = 0; mi < 4; ++mi)
#pragma unroll
            for (int ni = 0; ni < 2; ++ni) {
                acc[mi][ni] = __builtin_amdgcn_wmma_f32_16x16x32_bf16(
                    false, ahi[mi], false, bhi[ni], (short)0, acc[mi][ni], false, false);
                acc[mi][ni] = __builtin_amdgcn_wmma_f32_16x16x32_bf16(
                    false, ahi[mi], false, blo[ni], (short)0, acc[mi][ni], false, false);
                acc[mi][ni] = __builtin_amdgcn_wmma_f32_16x16x32_bf16(
                    false, alo[mi], false, bhi[ni], (short)0, acc[mi][ni], false, false);
            }
        __syncthreads();  // next iteration's issue() overwrites the other buffer
    }

    // ---- epilogue: C 16x16 f32 layout (VGPR r: M=r / M=r+8 for lane halves) ----
#pragma unroll
    for (int mi = 0; mi < 4; ++mi) {
#pragma unroll
        for (int ni = 0; ni < 2; ++ni) {
            const int n  = nBase + waveN * 32 + ni * 16 + lrow;
            const float bv = bias[n];
            const int m0 = mBase + waveM * 64 + mi * 16 + (hiHalf ? 8 : 0);
#pragma unroll
            for (int rr = 0; rr < 8; ++rr)
                out[(size_t)(m0 + rr) * OUT_F + n] = acc[mi][ni][rr] + bv;
        }
    }
}

extern "C" void kernel_launch(void* const* d_in, const int* in_sizes, int n_in,
                              void* d_out, int out_size, void* d_ws, size_t ws_size,
                              hipStream_t stream) {
    const float* x   = (const float*)d_in[0];
    const float* wv  = (const float*)d_in[1];
    const int*   wr  = (const int*)  d_in[2];
    const int*   wc  = (const int*)  d_in[3];
    const float* bv  = (const float*)d_in[4];
    const int*   bi  = (const int*)  d_in[5];
    const float* e0v = (const float*)d_in[6];
    const int*   e0p = (const int*)  d_in[7];
    const float* e1v = (const float*)d_in[8];
    const int*   e1p = (const int*)  d_in[9];
    float* out = (float*)d_out;

    // Workspace: W f32 | bias | xe | Xhi | Xlo | Whi | Wlo   (~106 MB)
    constexpr size_t W_ELEMS = (size_t)OUT_F * IN_TOT;   // 4,456,448
    constexpr size_t X_ELEMS = (size_t)BATCH * IN_TOT;   // 17,825,792
    float*  W   = (float*)d_ws;
    float*  b   = W + W_ELEMS;
    float*  xe  = b + OUT_F;
    __bf16* Xhi = (__bf16*)(xe + (size_t)BATCH * 2 * N_EMB);
    __bf16* Xlo = Xhi + X_ELEMS;
    __bf16* Whi = Xlo + X_ELEMS;
    __bf16* Wlo = Whi + W_ELEMS;

    const int nzw  = in_sizes[1];
    const int nzb  = in_sizes[4];
    const int ztot = OUT_F * IN_TOT + OUT_F;  // W and b are contiguous

    zero_kernel     <<<1024, 256, 0, stream>>>(W, ztot);
    scatter_w_kernel<<<(nzw + 255) / 256, 256, 0, stream>>>(W, wv, wr, wc, nzw);
    scatter_b_kernel<<<(nzb + 255) / 256, 256, 0, stream>>>(b, bv, bi, nzb);
    embed_kernel    <<<BATCH / 256, 256, 0, stream>>>(x, xe, e0v, e0p, e1v, e1p);
    split_x_kernel  <<<(BATCH * (IN_TOT / 4)) / 256, 256, 0, stream>>>(x, xe, Xhi, Xlo);
    split_w_kernel  <<<(OUT_F * (IN_TOT / 4)) / 256, 256, 0, stream>>>(W, Whi, Wlo);
    gemm_kernel     <<<dim3(OUT_F / TILE_N, BATCH / TILE_M), 256, 0, stream>>>(
        Xhi, Xlo, Whi, Wlo, b, out);
}
